// AttentiveRNNLanguageModel_60997125538239
// MI455X (gfx1250) — matrix operand order repaired and checked by reading, not compiled, not run on hardware
//
#include <hip/hip_runtime.h>
#include <hip/hip_bf16.h>
#include <math.h>

#define T_SEQ 512
#define B_SZ  8
#define H_SZ  512
#define P_SZ  20
#define V_SZ  32000
#define MT    (B_SZ * T_SEQ)   /* 4096 rows of (b,t) */

typedef __attribute__((ext_vector_type(16))) __bf16 v16bf;
typedef __attribute__((ext_vector_type(8)))  float  v8f;
typedef unsigned int v4u __attribute__((ext_vector_type(4)));
typedef int          v4i __attribute__((ext_vector_type(4)));
typedef int          v8i __attribute__((ext_vector_type(8)));

__device__ __forceinline__ float sigf(float x) { return 1.0f / (1.0f + __expf(-x)); }

/* LDS byte offset of a __shared__ object (generic -> addrspace(3) -> int) */
__device__ __forceinline__ unsigned lds_off_of(const void* p) {
  return (unsigned)(unsigned long long)(const __attribute__((address_space(3))) char*)p;
}

/* ------------------- Tensor Data Mover: 2D tile -> LDS ---------------------
   D# per CDNA5 ISA ch.8: group0 {count=1, lds_addr, global_addr[56:0], type=2},
   group1 {data_size, tensor_dim0/1 (OOB rows/cols read as zero), tile_dim0/1,
   tensor_dim0_stride}. bf16 => data_size code 1 (2 bytes). Dims/strides in
   element units. Groups 2/3 zero (<=2D). Tracked by TENSORcnt.
   This toolchain exposes the 6-arg builtin (extra zero group + cpol).       */
__device__ __forceinline__ void tdm_load_2d(const void* gptr, unsigned lds_byte_off,
                                            unsigned tile_w, unsigned tile_h,
                                            unsigned tensor_w, unsigned tensor_h,
                                            unsigned long long stride_elems) {
  unsigned long long ga = (unsigned long long)gptr;
  v4u g0;
  g0[0] = 1u;                                             /* count=1, user mode */
  g0[1] = lds_byte_off;                                   /* lds_addr           */
  g0[2] = (unsigned)ga;                                   /* global_addr[31:0]  */
  g0[3] = (unsigned)((ga >> 32) & 0x01FFFFFFu) | (2u << 30); /* [56:32] | type=2 */
  unsigned d0 = 1u << 16;                                 /* data_size=1 (2B)   */
  unsigned d1 = (tensor_w & 0xFFFFu) << 16;               /* tensor_dim0 lo     */
  unsigned d2 = (tensor_w >> 16) | ((tensor_h & 0xFFFFu) << 16);
  unsigned d3 = (tensor_h >> 16) | (tile_w << 16);        /* tile_dim0          */
  unsigned d4 = tile_h & 0xFFFFu;                         /* tile_dim1          */
  unsigned d5 = (unsigned)stride_elems;                   /* dim0_stride lo     */
  unsigned d6 = (unsigned)(stride_elems >> 32) & 0xFFFFu; /* dim0_stride hi     */
  v8i g1 = {(int)d0, (int)d1, (int)d2, (int)d3, (int)d4, (int)d5, (int)d6, 0};
  v4i z4 = {0, 0, 0, 0};
  v8i z8 = {0, 0, 0, 0, 0, 0, 0, 0};
  __builtin_amdgcn_tensor_load_to_lds(g0, g1, z4, z4, z8, 0);
}

/* ----------------------------- utility kernels ----------------------------- */

__global__ void k_cvt_bf16(const float* __restrict__ in, __bf16* __restrict__ out, long long n) {
  long long i = (long long)blockIdx.x * blockDim.x + threadIdx.x;
  long long stride = (long long)gridDim.x * blockDim.x;
  for (; i < n; i += stride) out[i] = (__bf16)in[i];
}

__global__ void k_embed(const int* __restrict__ tok, const float* __restrict__ emb,
                        __bf16* __restrict__ out) {
  int row = blockIdx.x;                       // b*T + t
  int id  = tok[row];
  const float* src = emb + (long long)id * H_SZ;
  __bf16*      dst = out + (long long)row * H_SZ;
  for (int k = threadIdx.x; k < H_SZ; k += blockDim.x) dst[k] = (__bf16)src[k];
}

__global__ void k_init(__bf16* __restrict__ h, unsigned* __restrict__ cnt) {
  for (int k = threadIdx.x; k < 16 * H_SZ; k += blockDim.x) h[k] = (__bf16)0.0f;
  if (threadIdx.x == 0) *cnt = 0u;
}

/* ------------------------ generic bf16 WMMA GEMM --------------------------- */
/* C[M,N] = A[M,K] * op(B) (+bias0+bias1) (optional tanh)
   bT=1: B stored [N,K] (A*B^T), both tiles DMA'd by the TDM.
   bT=0: B stored [K,N] (A*B), B staged manually (transpose), A via TDM.
   Batched via blockIdx.z. Output f32 and/or bf16 with leading dim ldo.      */

#define BM 64
#define BN 64
#define BK 32

__launch_bounds__(256)
__global__ void k_wmma_gemm(const __bf16* __restrict__ A, long long lda, long long strideA,
                            const __bf16* __restrict__ B, long long ldb, long long strideB,
                            float* outF, __bf16* outB, long long ldo, long long strideO,
                            const float* bias0, const float* bias1,
                            int M, int N, int K, int act, int bT) {
  __shared__ __bf16 sA[BM][BK];
  __shared__ __bf16 sB[BN][BK];

  const int z = blockIdx.z;
  A += (long long)z * strideA;
  B += (long long)z * strideB;
  const long long obase = (long long)z * strideO;

  const int m0 = blockIdx.y * BM, n0 = blockIdx.x * BN;
  const int tid  = threadIdx.x;
  const int w    = tid >> 5, lane = tid & 31;
  const int half = lane >> 4, ln = lane & 15;
  const int tm = w & 3;             // 0..3  (M sub-tile)
  const int tn = (w >> 2) << 1;     // 0 or 2 (N sub-tile pair)

  const unsigned sA_off = lds_off_of(&sA[0][0]);
  const unsigned sB_off = lds_off_of(&sB[0][0]);

  v8f acc0 = {0, 0, 0, 0, 0, 0, 0, 0};
  v8f acc1 = {0, 0, 0, 0, 0, 0, 0, 0};

  for (int kb = 0; kb < K; kb += BK) {
    /* --- stage A tile (64x32) via Tensor Data Mover; OOB rows -> zeros --- */
    if (w == 0) {
      tdm_load_2d(A + (long long)m0 * lda + kb, sA_off, BK, BM,
                  (unsigned)(K - kb), (unsigned)(M - m0), (unsigned long long)lda);
      if (bT) {
        tdm_load_2d(B + (long long)n0 * ldb + kb, sB_off, BK, BN,
                    (unsigned)(K - kb), (unsigned)(N - n0), (unsigned long long)ldb);
      }
      __builtin_amdgcn_s_wait_tensorcnt(0);
    }
    /* --- B tile for A*B form: manual transposed staging [K,N] -> [n][k] --- */
    if (!bT) {
      int idx = tid * 8, k = idx >> 6, n = idx & 63;
      const __bf16* s = B + (long long)(kb + k) * ldb + n0 + n;
#pragma unroll
      for (int i = 0; i < 8; i++)
        sB[n + i][k] = (n0 + n + i < N) ? s[i] : (__bf16)0.0f;
      if (kb + BK < K) __builtin_prefetch(s + (long long)BK * ldb, 0, 1);
    }
    __syncthreads();

    /* --- fragments: A and B both use contiguous k = 16*half + i --- */
    v16bf af, bf0, bf1;
    {
      int m = (tm << 4) + ln;
#pragma unroll
      for (int i = 0; i < 8; i++) af[i] = sA[m][(half << 4) + i];
#pragma unroll
      for (int i = 0; i < 8; i++) af[8 + i] = sA[m][(half << 4) + 8 + i];
    }
    {
      int n = (tn << 4) + ln;
#pragma unroll
      for (int i = 0; i < 16; i++) bf0[i] = sB[n][(half << 4) + i];
    }
    {
      int n = ((tn + 1) << 4) + ln;
#pragma unroll
      for (int i = 0; i < 16; i++) bf1[i] = sB[n][(half << 4) + i];
    }
    acc0 = __builtin_amdgcn_wmma_f32_16x16x32_bf16(false, af, false, bf0, (short)0, acc0, false, false);
    acc1 = __builtin_amdgcn_wmma_f32_16x16x32_bf16(false, af, false, bf1, (short)0, acc1, false, false);
    __syncthreads();
  }

  /* --- epilogue: C layout lane=column, vgpr v -> row v+8*half --- */
#pragma unroll
  for (int v = 0; v < 8; v++) {
    int mrow = m0 + (tm << 4) + v + (half << 3);
    if (mrow >= M) continue;
#pragma unroll
    for (int q = 0; q < 2; q++) {
      int ncol = n0 + ((tn + q) << 4) + ln;
      if (ncol >= N) continue;
      float val = q ? acc1[v] : acc0[v];
      if (bias0) val += bias0[ncol];
      if (bias1) val += bias1[ncol];
      if (act)   val = tanhf(val);
      long long o = obase + (long long)mrow * ldo + ncol;
      if (outF) outF[o] = val;
      if (outB) outB[o] = (__bf16)val;
    }
  }
}

/* -------------------- encoder LSTM (persistent, WMMA) ---------------------- */
/* 16 blocks; block k owns hidden cols [32k,32k+32): 4 gates x 32 rows = 8
   16x16 N-tiles, one per wave. h (16x512, rows 8..15 zero) DMA'd into LDS by
   the TDM each step, Whh bf16 streamed from L2. Cross-block sync: atomic
   grid barrier (counter reset by k_init each launch).                       */

#define NB_LSTM 16

__launch_bounds__(256)
__global__ void k_enc_lstm(const float* __restrict__ xg,    /* [MT, 2048] */
                           const __bf16* __restrict__ Whh,  /* [2048, 512] */
                           __bf16* __restrict__ hglob,      /* [16, 512]  */
                           __bf16* __restrict__ ctxenc,     /* [MT, 1024], right half */
                           unsigned* __restrict__ cnt) {
  __shared__ __bf16 sH[16][H_SZ];
  __shared__ float  sG[8][128];

  const int tid  = threadIdx.x;
  const int w    = tid >> 5, lane = tid & 31;
  const int half = lane >> 4, ln = lane & 15;
  const int jbase  = blockIdx.x * 32;
  const int gate   = w >> 1, colgrp = w & 1;
  const int nrow0  = gate * H_SZ + jbase + colgrp * 16;
  const int ub = tid >> 5;   /* update-phase batch 0..7 */
  const int uj = tid & 31;   /* update-phase column     */
  const unsigned sH_off = lds_off_of(&sH[0][0]);
  float creg = 0.0f;

  for (int t = 0; t < T_SEQ; ++t) {
    /* stage h into LDS via TDM (16 x 512 bf16 tile) */
    if (w == 0) {
      tdm_load_2d(hglob, sH_off, H_SZ, 16, H_SZ, 16, (unsigned long long)H_SZ);
      __builtin_amdgcn_s_wait_tensorcnt(0);
    }
    __syncthreads();

    v8f acc;
#pragma unroll
    for (int v = 0; v < 8; v++) {
      int m = v + (half << 3);
      acc[v] = (m < 8) ? xg[((long long)m * T_SEQ + t) * 2048 + nrow0 + ln] : 0.0f;
    }
    for (int kb = 0; kb < H_SZ; kb += 32) {
      v16bf af, bf;
#pragma unroll
      for (int i = 0; i < 8; i++) af[i] = sH[ln][kb + (half << 4) + i];
#pragma unroll
      for (int i = 0; i < 8; i++) af[8 + i] = sH[ln][kb + (half << 4) + 8 + i];
      const __bf16* bp = Whh + (long long)(nrow0 + ln) * H_SZ + kb + (half << 4);
#pragma unroll
      for (int i = 0; i < 16; i++) bf[i] = bp[i];
      acc = __builtin_amdgcn_wmma_f32_16x16x32_bf16(false, af, false, bf, (short)0, acc, false, false);
    }
#pragma unroll
    for (int v = 0; v < 8; v++) {
      int m = v + (half << 3);
      if (m < 8) sG[m][gate * 32 + colgrp * 16 + ln] = acc[v];
    }
    __syncthreads();

    /* gate math: thread -> fixed (b, j), cell state lives in a register */
    {
      float gi = sG[ub][uj], gf = sG[ub][32 + uj], gg = sG[ub][64 + uj], go = sG[ub][96 + uj];
      creg = sigf(gf) * creg + sigf(gi) * tanhf(gg);
      float hv = sigf(go) * tanhf(creg);
      __bf16 hb = (__bf16)hv;
      hglob[ub * H_SZ + jbase + uj] = hb;
      ctxenc[((long long)ub * T_SEQ + t) * 1024 + 512 + jbase + uj] = hb;
    }

    /* grid barrier (monotonic counter; reset by k_init each launch) */
    __syncthreads();
    if (tid == 0) {
      __threadfence();
      atomicAdd(cnt, 1u);
      volatile unsigned* vc = (volatile unsigned*)cnt;
      unsigned target = (unsigned)NB_LSTM * (unsigned)(t + 1);
      while (*vc < target) __builtin_amdgcn_s_sleep(1);
      __threadfence();
    }
    __syncthreads();
  }
}

/* ------------------------------ pos LSTM (P=20) ---------------------------- */

__launch_bounds__(256)
__global__ void k_pos_lstm(const float* __restrict__ xg,   /* [MT, 80] */
                           const float* __restrict__ Whh,  /* [80, 20] */
                           float* __restrict__ pos) {      /* [MT, 20] */
  __shared__ float sW[80][20];
  __shared__ float sh[8][20];
  __shared__ float sgt[8][80];
  int tid = threadIdx.x;
  for (int i = tid; i < 80 * 20; i += 256) sW[i / 20][i % 20] = Whh[i];
  for (int i = tid; i < 8 * 20; i += 256)  sh[i / 20][i % 20] = 0.0f;
  __syncthreads();
  float creg = 0.0f;
  int ub = tid / 20, uj = tid % 20;
  for (int t = 0; t < T_SEQ; ++t) {
    for (int i = tid; i < 640; i += 256) {
      int b = i / 80, r = i % 80;
      float s = xg[((long long)b * T_SEQ + t) * 80 + r];
      for (int k = 0; k < P_SZ; k++) s += sh[b][k] * sW[r][k];
      sgt[b][r] = s;
    }
    __syncthreads();
    if (tid < 160) {
      float gi = sgt[ub][uj], gf = sgt[ub][20 + uj], gg = sgt[ub][40 + uj], go = sgt[ub][60 + uj];
      creg = sigf(gf) * creg + sigf(gi) * tanhf(gg);
      float hv = sigf(go) * tanhf(creg);
      sh[ub][uj] = hv;
      pos[((long long)ub * T_SEQ + t) * P_SZ + uj] = hv;
    }
    __syncthreads();
  }
}

/* ----------------------- mu/sigma scan (per batch) ------------------------- */

__global__ void k_mu(const float* __restrict__ pos, const int* __restrict__ pl,
                     const float* __restrict__ Wmu, const float* __restrict__ bmu,
                     const float* __restrict__ Wsig, const float* __restrict__ bsig,
                     float* __restrict__ mu, float* __restrict__ sg) {
  int b = threadIdx.x;
  if (b >= B_SZ) return;
  float L = (float)pl[b];
  float prev = 0.0f;
  for (int t = 0; t < T_SEQ; ++t) {
    const float* p = pos + ((long long)b * T_SEQ + t) * P_SZ;
    float m0 = bmu[0], m1 = bmu[1], m2 = bmu[2], ss = bsig[0];
    for (int k = 0; k < P_SZ; k++) {
      float pv = p[k];
      m0 += Wmu[0 * P_SZ + k] * pv;
      m1 += Wmu[1 * P_SZ + k] * pv;
      m2 += Wmu[2 * P_SZ + k] * pv;
      ss += Wsig[k] * pv;
    }
    m0 = fmaxf(m0, 0.0f); m1 = fmaxf(m1, 0.0f); m2 = fmaxf(m2, 0.0f);
    ss = sigf(ss);
    float muv = m0 * prev + m1 / L + m2 * ((float)(t + 1)) / L;
    muv = fmaxf(muv, (float)t / L);
    mu[b * T_SEQ + t] = muv;
    sg[b * T_SEQ + t] = ss;
    prev = muv;
  }
}

/* -------------------- Gaussian prefix weights (bf16 g) --------------------- */

__launch_bounds__(256)
__global__ void k_gw(const float* __restrict__ mu, const float* __restrict__ sg,
                     const int* __restrict__ pl, __bf16* __restrict__ g) {
  __shared__ float red[256];
  int tq = blockIdx.x, b = blockIdx.y, tid = threadIdx.x;
  float L = (float)pl[b];
  float muv = mu[b * T_SEQ + tq], sv = sg[b * T_SEQ + tq];
  float denom = 2.0f * sv * sv + 0.001f;
  float ws[2];
  float psum = 0.0f;
#pragma unroll
  for (int q = 0; q < 2; q++) {
    int tk = tid + q * 256;
    float d = (float)tk / L - muv;
    float wv = (tk <= tq) ? __expf(-d * d / denom) : 0.0f;
    ws[q] = wv;
    psum += wv;
  }
  red[tid] = psum;
  __syncthreads();
  for (int s = 128; s > 0; s >>= 1) {
    if (tid < s) red[tid] += red[tid + s];
    __syncthreads();
  }
  float inv = 1.0f / fmaxf(red[0], 1e-12f);
  __bf16* dst = g + ((long long)b * T_SEQ + tq) * T_SEQ;
#pragma unroll
  for (int q = 0; q < 2; q++) dst[tid + q * 256] = (__bf16)(ws[q] * inv);
}

/* --------------------------------- host ------------------------------------ */

extern "C" void kernel_launch(void* const* d_in, const int* in_sizes, int n_in,
                              void* d_out, int out_size, void* d_ws, size_t ws_size,
                              hipStream_t stream) {
  (void)in_sizes; (void)n_in; (void)out_size; (void)ws_size;
  const int*   tokens = (const int*)d_in[0];
  const int*   plen   = (const int*)d_in[1];
  const float* emb    = (const float*)d_in[2];
  const float* eWih   = (const float*)d_in[3];
  const float* eWhh   = (const float*)d_in[4];
  const float* ebih   = (const float*)d_in[5];
  const float* ebhh   = (const float*)d_in[6];
  const float* pWih   = (const float*)d_in[7];
  const float* pWhh   = (const float*)d_in[8];
  const float* pbih   = (const float*)d_in[9];
  const float* pbhh   = (const float*)d_in[10];
  const float* Wmu    = (const float*)d_in[11];
  const float* bmu    = (const float*)d_in[12];
  const float* Wsig   = (const float*)d_in[13];
  const float* bsig   = (const float*)d_in[14];
  const float* Wcat   = (const float*)d_in[15];
  const float* bcat   = (const float*)d_in[16];
  const float* decb   = (const float*)d_in[17];
  float* out = (float*)d_out;

  char* ws = (char*)d_ws;
  size_t off = 0;
  auto alloc = [&](size_t bytes) -> char* {
    char* p = ws + off;
    off = (off + bytes + 255) & ~(size_t)255;
    return p;
  };

  __bf16* emb_bf    = (__bf16*)alloc((size_t)MT * H_SZ * 2);
  __bf16* WihE_bf   = (__bf16*)alloc((size_t)2048 * H_SZ * 2);
  __bf16* WhhE_bf   = (__bf16*)alloc((size_t)2048 * H_SZ * 2);
  __bf16* pWih_bf   = (__bf16*)alloc((size_t)80 * H_SZ * 2);
  __bf16* Wcat_bf   = (__bf16*)alloc((size_t)H_SZ * 1024 * 2);
  __bf16* embT_bf   = (__bf16*)alloc((size_t)V_SZ * H_SZ * 2);
  float*  xg_enc    = (float*)alloc((size_t)MT * 2048 * 4);
  __bf16* ctxenc    = (__bf16*)alloc((size_t)MT * 1024 * 2);  /* [ctx | enc] */
  __bf16* h_bf      = (__bf16*)alloc((size_t)16 * H_SZ * 2);
  float*  xg_pos    = (float*)alloc((size_t)MT * 80 * 4);
  float*  pos       = (float*)alloc((size_t)MT * P_SZ * 4);
  float*  mu        = (float*)alloc((size_t)MT * 4);
  float*  sgv       = (float*)alloc((size_t)MT * 4);
  __bf16* g_bf      = (__bf16*)alloc((size_t)B_SZ * T_SEQ * T_SEQ * 2);
  __bf16* comb_bf   = (__bf16*)alloc((size_t)MT * H_SZ * 2);
  unsigned* counter = (unsigned*)alloc(256);

  auto gemm = [&](const __bf16* A, long long lda, long long sA,
                  const __bf16* B, long long ldb, long long sB,
                  float* oF, __bf16* oB, long long ldo, long long sO,
                  const float* b0, const float* b1,
                  int M, int N, int K, int act, int bT, int batch) {
    dim3 grid((N + BN - 1) / BN, (M + BM - 1) / BM, batch);
    k_wmma_gemm<<<grid, dim3(256), 0, stream>>>(A, lda, sA, B, ldb, sB, oF, oB, ldo, sO,
                                                b0, b1, M, N, K, act, bT);
  };
  auto cvt = [&](const float* in, __bf16* o, long long n) {
    int blocks = (int)((n + 2047) / 2048);
    k_cvt_bf16<<<dim3(blocks), dim3(256), 0, stream>>>(in, o, n);
  };

  /* 0: reset recurrent state + barrier counter (deterministic each launch) */
  k_init<<<dim3(1), dim3(256), 0, stream>>>(h_bf, counter);

  /* 1: weight / table converts to bf16 */
  cvt(eWih, WihE_bf, (long long)2048 * H_SZ);
  cvt(eWhh, WhhE_bf, (long long)2048 * H_SZ);
  cvt(pWih, pWih_bf, (long long)80 * H_SZ);
  cvt(Wcat, Wcat_bf, (long long)H_SZ * 1024);
  cvt(emb,  embT_bf, (long long)V_SZ * H_SZ);

  /* 2: embedding gather -> bf16 */
  k_embed<<<dim3(MT), dim3(256), 0, stream>>>(tokens, emb, emb_bf);

  /* 3: xg_enc = emb @ enc_Wih^T + bih + bhh   [4096,2048] */
  gemm(emb_bf, H_SZ, 0, WihE_bf, H_SZ, 0, xg_enc, nullptr, 2048, 0,
       ebih, ebhh, MT, 2048, H_SZ, 0, 1, 1);

  /* 4: encoder LSTM (writes enc bf16 into right half of ctxenc) */
  k_enc_lstm<<<dim3(NB_LSTM), dim3(256), 0, stream>>>(xg_enc, WhhE_bf, h_bf, ctxenc, counter);

  /* 5: xg_pos = enc @ pos_Wih^T + pbih + pbhh   [4096,80] */
  gemm(ctxenc + 512, 1024, 0, pWih_bf, H_SZ, 0, xg_pos, nullptr, 80, 0,
       pbih, pbhh, MT, 80, H_SZ, 0, 1, 1);

  /* 6: pos LSTM, 7: mu/sigma scan, 8: Gaussian weights */
  k_pos_lstm<<<dim3(1), dim3(256), 0, stream>>>(xg_pos, pWhh, pos);
  k_mu<<<dim3(1), dim3(32), 0, stream>>>(pos, plen, Wmu, bmu, Wsig, bsig, mu, sgv);
  k_gw<<<dim3(T_SEQ, B_SZ), dim3(256), 0, stream>>>(mu, sgv, plen, g_bf);

  /* 9: ctx = g @ enc (batched, A*B form), write bf16 into left half of ctxenc */
  gemm(g_bf, T_SEQ, (long long)T_SEQ * T_SEQ,
       ctxenc + 512, 1024, (long long)T_SEQ * 1024,
       nullptr, ctxenc, 1024, (long long)T_SEQ * 1024,
       nullptr, nullptr, T_SEQ, H_SZ, T_SEQ, 0, 0, B_SZ);

  /* 10: comb = tanh([ctx|enc] @ W_cat^T + b_cat) -> bf16 */
  gemm(ctxenc, 1024, 0, Wcat_bf, 1024, 0, nullptr, comb_bf, H_SZ, 0,
       bcat, nullptr, MT, H_SZ, 1024, 1, 1, 1);

  /* 11: logits = comb @ embedding^T + dec_b  (the 134 GFLOP decoder) */
  gemm(comb_bf, H_SZ, 0, embT_bf, H_SZ, 0, out, nullptr, V_SZ, 0,
       decb, nullptr, MT, V_SZ, H_SZ, 0, 1, 1);
}
